// MultiHeadAttention_16509854286024
// MI455X (gfx1250) — matrix operand order repaired and checked by reading, not compiled, not run on hardware
//
#include <hip/hip_runtime.h>

#define BATCH 4
#define SEQ   2048
#define EMB   1024
#define HEADS 16
#define DHEAD 64

typedef __bf16 bf16_t;
typedef __attribute__((ext_vector_type(16))) bf16_t v16bf;
typedef __attribute__((ext_vector_type(8)))  float  v8f;
typedef __attribute__((ext_vector_type(4)))  unsigned int u32x4;

union Frag16 {
  v16bf v;
  unsigned short u[16];
  u32x4 q[2];
};

__device__ __forceinline__ unsigned short f2bf(float f) {
  unsigned int u = __builtin_bit_cast(unsigned int, f);
  unsigned int r = 0x7FFFu + ((u >> 16) & 1u);
  return (unsigned short)((u + r) >> 16);
}

__device__ __forceinline__ v8f vzero() {
  v8f z;
#pragma unroll
  for (int i = 0; i < 8; ++i) z[i] = 0.0f;
  return z;
}

__device__ __forceinline__ v8f wmma_bf16(Frag16 a, Frag16 b, v8f c) {
  // D = A(16x32 bf16) * B(32x16 bf16) + C(16x16 f32)
  return __builtin_amdgcn_wmma_f32_16x16x32_bf16(
      /*neg_a=*/false, a.v, /*neg_b=*/false, b.v,
      /*c_mod=*/(short)0, c, /*reuse_a=*/false, /*reuse_b=*/false);
}

// A fragment, row-major source [rows][ld] (bf16 as ushort).
// Lane layout: row = rowbase + lane%16;
//   lanes 0-15 : K = kb+{0..7, 16..23}; lanes 16-31 : K = kb+{8..15, 24..31}
__device__ __forceinline__ Frag16 load_a(const unsigned short* src, int ld,
                                         int row, int kb, int lane) {
  Frag16 f;
  int hi = lane >> 4;
  const unsigned short* p = src + (size_t)row * ld + kb + hi * 8;
  f.q[0] = *(const u32x4*)(p);
  f.q[1] = *(const u32x4*)(p + 16);
  return f;
}

// B fragment, source laid out column-contiguous: element (k,n) at src[n*ld + k].
// Lane layout: col n; lanes 0-15 : K = kb+{0..7, 16..23}; lanes 16-31 : K = kb+{8..15, 24..31}
__device__ __forceinline__ Frag16 load_b(const unsigned short* src, int ld,
                                         int n, int kb, int lane) {
  Frag16 f;
  int hi = lane >> 4;
  const unsigned short* p = src + (size_t)n * ld + kb + hi * 8;
  f.q[0] = *(const u32x4*)(p);
  f.q[1] = *(const u32x4*)(p + 16);
  return f;
}

// A fragment from a 16x32 bf16 tile staged in LDS (row-major, ld=32)
__device__ __forceinline__ Frag16 load_a_lds(const unsigned short* L, int lane) {
  Frag16 f;
  int row = lane & 15, hi = lane >> 4;
  const unsigned short* p = L + row * 32 + hi * 8;
  f.q[0] = *(const u32x4*)(p);
  f.q[1] = *(const u32x4*)(p + 16);
  return f;
}

__device__ __forceinline__ float rmax16(float x) {
#pragma unroll
  for (int m = 1; m < 16; m <<= 1) x = fmaxf(x, __shfl_xor(x, m, 32));
  return x;
}
__device__ __forceinline__ float rsum16(float x) {
#pragma unroll
  for (int m = 1; m < 16; m <<= 1) x += __shfl_xor(x, m, 32);
  return x;
}

// ---------------- stage 0: conversions ----------------

__global__ __launch_bounds__(256) void cvt_f32_bf16(const float* __restrict__ in,
                                                    unsigned short* __restrict__ out,
                                                    int n) {
  int i = blockIdx.x * 256 + threadIdx.x;
  if (i < n) out[i] = f2bf(in[i]);
}

// W[h][e][d] (f32) -> Wt[h][d][e] (bf16)  (column-contiguous for B fragments)
__global__ __launch_bounds__(256) void cvt_w_transpose(const float* __restrict__ W,
                                                       unsigned short* __restrict__ Wt) {
  int i = blockIdx.x * 256 + threadIdx.x;   // i over H*DH*E = 1M
  int e = i & (EMB - 1);
  int d = (i >> 10) & (DHEAD - 1);
  int h = i >> 16;
  Wt[i] = f2bf(W[((size_t)(h * EMB + e)) * DHEAD + d]);
}

// ---------------- stage 1: QKV projection ----------------
// One wave -> 32 rows x 64 cols (full head) for one of {Q,K,V}.
// Double-buffered K-loop. V is written TRANSPOSED [B,H,DH,T] so the flash
// kernel's P*V B-fragments are contiguous b128 loads.

__global__ __launch_bounds__(128) void qkv_kernel(
    const unsigned short* __restrict__ xb,
    const unsigned short* __restrict__ Wqt, const unsigned short* __restrict__ Wkt,
    const unsigned short* __restrict__ Wvt,
    unsigned short* __restrict__ Qb, unsigned short* __restrict__ Kb,
    unsigned short* __restrict__ Vt) {
  int lane = threadIdx.x & 31;
  int gw = blockIdx.x * 4 + (threadIdx.x >> 5);
  int m  = gw >> 12;                 // 4096 waves per matrix
  int rr = gw & 4095;
  int rt = rr >> 4;                  // 256 tiles of 32 rows over B*T
  int h  = rr & (HEADS - 1);

  const unsigned short* W = (m == 0 ? Wqt : (m == 1 ? Wkt : Wvt)) + (size_t)h * DHEAD * EMB;

  int col = lane & 15, hi = lane >> 4;
  int rowbase = rt * 32;

  v8f acc[2][4];
#pragma unroll
  for (int s = 0; s < 2; ++s)
#pragma unroll
    for (int j = 0; j < 4; ++j) acc[s][j] = vzero();

  Frag16 a0 = load_a(xb, EMB, rowbase + col, 0, lane);
  Frag16 a1 = load_a(xb, EMB, rowbase + 16 + col, 0, lane);
  Frag16 bfr[4];
#pragma unroll
  for (int j = 0; j < 4; ++j) bfr[j] = load_b(W, EMB, j * 16 + col, 0, lane);

  for (int kb = 0; kb < EMB - 32; kb += 32) {
    // prefetch next K-step while computing current
    Frag16 a0n = load_a(xb, EMB, rowbase + col, kb + 32, lane);
    Frag16 a1n = load_a(xb, EMB, rowbase + 16 + col, kb + 32, lane);
    Frag16 bn[4];
#pragma unroll
    for (int j = 0; j < 4; ++j) bn[j] = load_b(W, EMB, j * 16 + col, kb + 32, lane);

#pragma unroll
    for (int j = 0; j < 4; ++j) acc[0][j] = wmma_bf16(a0, bfr[j], acc[0][j]);
#pragma unroll
    for (int j = 0; j < 4; ++j) acc[1][j] = wmma_bf16(a1, bfr[j], acc[1][j]);

    a0 = a0n; a1 = a1n;
#pragma unroll
    for (int j = 0; j < 4; ++j) bfr[j] = bn[j];
  }
#pragma unroll
  for (int j = 0; j < 4; ++j) acc[0][j] = wmma_bf16(a0, bfr[j], acc[0][j]);
#pragma unroll
  for (int j = 0; j < 4; ++j) acc[1][j] = wmma_bf16(a1, bfr[j], acc[1][j]);

  if (m != 2) {
    unsigned short* O = (m == 0 ? Qb : Kb);   // [B,H,T,DH]
#pragma unroll
    for (int s = 0; s < 2; ++s)
#pragma unroll
      for (int j = 0; j < 4; ++j)
#pragma unroll
        for (int r = 0; r < 8; ++r) {
          int row = rowbase + s * 16 + r + 8 * hi;   // flat over B*T
          int t = row & (SEQ - 1);
          int b = row >> 11;
          int d = j * 16 + col;
          O[((size_t)((b * HEADS + h) * SEQ + t)) * DHEAD + d] = f2bf(acc[s][j][r]);
        }
  } else {
    // V transposed: Vt[b][h][d][t]
#pragma unroll
    for (int s = 0; s < 2; ++s)
#pragma unroll
      for (int j = 0; j < 4; ++j)
#pragma unroll
        for (int r = 0; r < 8; ++r) {
          int row = rowbase + s * 16 + r + 8 * hi;
          int t = row & (SEQ - 1);
          int b = row >> 11;
          int d = j * 16 + col;
          Vt[((size_t)((b * HEADS + h) * DHEAD + d)) * SEQ + t] = f2bf(acc[s][j][r]);
        }
  }
}

// ---------------- stage 2: causal flash attention ----------------
// One wave -> one (b, h, 16-query tile). Key blocks of 32, software-pipelined:
// block i+1's K/V fragments are loaded while block i's softmax + P*V run.

__global__ __launch_bounds__(128) void flash_kernel(
    const unsigned short* __restrict__ Qb, const unsigned short* __restrict__ Kb,
    const unsigned short* __restrict__ Vt, unsigned short* __restrict__ OH) {
  __shared__ unsigned short lds[4][16 * 32];
  int lane = threadIdx.x & 31;
  int wv = threadIdx.x >> 5;
  int gw = blockIdx.x * 4 + wv;
  int qt = gw & (SEQ / 16 - 1);         // 128 query tiles
  int h  = (gw >> 7) & (HEADS - 1);
  int b  = gw >> 11;

  const size_t hbase = (size_t)(b * HEADS + h) * SEQ * DHEAD;
  const unsigned short* Q = Qb + hbase;
  const unsigned short* K = Kb + hbase;
  const unsigned short* V = Vt + hbase;   // [DH][T]
  unsigned short* L = lds[wv];

  int col = lane & 15, hi = lane >> 4;
  int qbase = qt * 16;
  int kend = qbase + 15;

  Frag16 qa0 = load_a(Q, DHEAD, qbase + col, 0, lane);
  Frag16 qa1 = load_a(Q, DHEAD, qbase + col, 32, lane);

  v8f o[4];
#pragma unroll
  for (int j = 0; j < 4; ++j) o[j] = vzero();
  float mrow[8], lrow[8];
#pragma unroll
  for (int r = 0; r < 8; ++r) { mrow[r] = -1e30f; lrow[r] = 0.0f; }

  const float scale = 0.125f;  // 1/sqrt(64)

  auto load_kv = [&](int kb, Frag16 vf[4], Frag16 kf[4]) {
#pragma unroll
    for (int j = 0; j < 4; ++j) vf[j] = load_b(V, SEQ, j * 16 + col, kb, lane);
    kf[0] = load_b(K, DHEAD, kb + col,      0,  lane);
    kf[1] = load_b(K, DHEAD, kb + col,      32, lane);
    kf[2] = load_b(K, DHEAD, kb + 16 + col, 0,  lane);
    kf[3] = load_b(K, DHEAD, kb + 16 + col, 32, lane);
  };

  Frag16 vb[4], kf[4];
  load_kv(0, vb, kf);

  for (int kb = 0; kb <= kend; kb += 32) {
    // S = Q * K^T with current fragments
    v8f s0 = vzero(), s1 = vzero();
    s0 = wmma_bf16(qa0, kf[0], s0);
    s0 = wmma_bf16(qa1, kf[1], s0);
    s1 = wmma_bf16(qa0, kf[2], s1);
    s1 = wmma_bf16(qa1, kf[3], s1);

    // prefetch next key block: in flight during softmax + P staging
    Frag16 vbn[4], kfn[4];
    int kbn = (kb + 32 <= kend) ? kb + 32 : kb;   // clamp (last iter: L0 hit)
    load_kv(kbn, vbn, kfn);

    float p0[8], p1[8], alpha[8];
#pragma unroll
    for (int r = 0; r < 8; ++r) {
      int srow = qbase + r + 8 * hi;
      float x0 = s0[r] * scale; if (kb + col > srow)      x0 = -1e30f;
      float x1 = s1[r] * scale; if (kb + 16 + col > srow) x1 = -1e30f;
      float tmax = rmax16(fmaxf(x0, x1));
      float mn = fmaxf(mrow[r], tmax);
      float a = __expf(mrow[r] - mn);
      mrow[r] = mn;
      float e0 = __expf(x0 - mn);
      float e1 = __expf(x1 - mn);
      lrow[r] = lrow[r] * a + rsum16(e0 + e1);
      alpha[r] = a; p0[r] = e0; p1[r] = e1;
    }

#pragma unroll
    for (int j = 0; j < 4; ++j)
#pragma unroll
      for (int r = 0; r < 8; ++r) o[j][r] *= alpha[r];

    // stage P (C-layout -> A-layout) through LDS as bf16
#pragma unroll
    for (int r = 0; r < 8; ++r) {
      L[(r + 8 * hi) * 32 + col]      = f2bf(p0[r]);
      L[(r + 8 * hi) * 32 + 16 + col] = f2bf(p1[r]);
    }
    asm volatile("s_wait_dscnt 0" ::: "memory");  // wave-local LDS RAW fence
    Frag16 pa = load_a_lds(L, lane);

    // O += P * V  (V fragments already resident; ds_load->wmma dep makes the
    // compiler wait on the LDS reads, so no trailing fence is needed)
#pragma unroll
    for (int j = 0; j < 4; ++j) o[j] = wmma_bf16(pa, vb[j], o[j]);

#pragma unroll
    for (int j = 0; j < 4; ++j) { vb[j] = vbn[j]; kf[j] = kfn[j]; }
  }

  // epilogue: normalize and write head output [B*T][E] bf16
#pragma unroll
  for (int j = 0; j < 4; ++j) {
#pragma unroll
    for (int r = 0; r < 8; ++r) {
      int srow = qbase + r + 8 * hi;
      float val = o[j][r] / lrow[r];
      size_t rowflat = (size_t)b * SEQ + srow;
      OH[rowflat * EMB + h * DHEAD + j * 16 + col] = f2bf(val);
    }
  }
}

// ---------------- stage 3: output projection + bias ----------------
// One wave -> 32 rows x 64 cols of d_out (f32), double-buffered K-loop.

__global__ __launch_bounds__(256) void proj_kernel(
    const unsigned short* __restrict__ OH, const unsigned short* __restrict__ Wpb,
    const float* __restrict__ bp, float* __restrict__ out) {
  int lane = threadIdx.x & 31;
  int gw = blockIdx.x * 8 + (threadIdx.x >> 5);
  int rt = gw >> 4;             // 256 tiles of 32 rows
  int jt = gw & 15;             // 16 col groups of 64

  int col = lane & 15, hi = lane >> 4;
  int rowbase = rt * 32;

  v8f acc[2][4];
#pragma unroll
  for (int s = 0; s < 2; ++s)
#pragma unroll
    for (int j = 0; j < 4; ++j) acc[s][j] = vzero();

  Frag16 a0 = load_a(OH, EMB, rowbase + col, 0, lane);
  Frag16 a1 = load_a(OH, EMB, rowbase + 16 + col, 0, lane);
  Frag16 bfr[4];
#pragma unroll
  for (int j = 0; j < 4; ++j)
    bfr[j] = load_b(Wpb, EMB, jt * 64 + j * 16 + col, 0, lane);

  for (int kb = 0; kb < EMB - 32; kb += 32) {
    Frag16 a0n = load_a(OH, EMB, rowbase + col, kb + 32, lane);
    Frag16 a1n = load_a(OH, EMB, rowbase + 16 + col, kb + 32, lane);
    Frag16 bn[4];
#pragma unroll
    for (int j = 0; j < 4; ++j)
      bn[j] = load_b(Wpb, EMB, jt * 64 + j * 16 + col, kb + 32, lane);

#pragma unroll
    for (int j = 0; j < 4; ++j) acc[0][j] = wmma_bf16(a0, bfr[j], acc[0][j]);
#pragma unroll
    for (int j = 0; j < 4; ++j) acc[1][j] = wmma_bf16(a1, bfr[j], acc[1][j]);

    a0 = a0n; a1 = a1n;
#pragma unroll
    for (int j = 0; j < 4; ++j) bfr[j] = bn[j];
  }
#pragma unroll
  for (int j = 0; j < 4; ++j) acc[0][j] = wmma_bf16(a0, bfr[j], acc[0][j]);
#pragma unroll
  for (int j = 0; j < 4; ++j) acc[1][j] = wmma_bf16(a1, bfr[j], acc[1][j]);

#pragma unroll
  for (int s = 0; s < 2; ++s)
#pragma unroll
    for (int j = 0; j < 4; ++j)
#pragma unroll
      for (int r = 0; r < 8; ++r) {
        int row = rowbase + s * 16 + r + 8 * hi;
        int cl = jt * 64 + j * 16 + col;
        out[(size_t)row * EMB + cl] = acc[s][j][r] + bp[cl];
      }
}

// ---------------- host launcher ----------------

extern "C" void kernel_launch(void* const* d_in, const int* in_sizes, int n_in,
                              void* d_out, int out_size, void* d_ws, size_t ws_size,
                              hipStream_t stream) {
  const float* x  = (const float*)d_in[0];
  const float* Wq = (const float*)d_in[1];
  const float* Wk = (const float*)d_in[2];
  const float* Wv = (const float*)d_in[3];
  const float* Wp = (const float*)d_in[4];
  const float* bp = (const float*)d_in[5];
  float* out = (float*)d_out;

  const size_t NX = (size_t)BATCH * SEQ * EMB;           // 8M elems
  const size_t NW = (size_t)HEADS * EMB * DHEAD;         // 1M elems
  const size_t NP = (size_t)EMB * EMB;                   // 1M elems
  const size_t NQ = (size_t)BATCH * HEADS * SEQ * DHEAD; // 8M elems

  unsigned short* w   = (unsigned short*)d_ws;
  unsigned short* xb  = w;
  unsigned short* Wqt = xb + NX;
  unsigned short* Wkt = Wqt + NW;
  unsigned short* Wvt = Wkt + NW;
  unsigned short* Wpb = Wvt + NW;
  unsigned short* Qb  = Wpb + NP;
  unsigned short* Kb  = Qb + NQ;
  unsigned short* Vt  = Kb + NQ;   // transposed V: [B,H,DH,T]
  unsigned short* OH  = Vt + NQ;

  cvt_f32_bf16<<<(int)(NX / 256), 256, 0, stream>>>(x, xb, (int)NX);
  cvt_f32_bf16<<<(int)(NP / 256), 256, 0, stream>>>(Wp, Wpb, (int)NP);
  cvt_w_transpose<<<(int)(NW / 256), 256, 0, stream>>>(Wq, Wqt);
  cvt_w_transpose<<<(int)(NW / 256), 256, 0, stream>>>(Wk, Wkt);
  cvt_w_transpose<<<(int)(NW / 256), 256, 0, stream>>>(Wv, Wvt);

  // QKV: 3 * 256 row-tiles(32) * 16 heads = 12288 waves, 4 waves / block
  qkv_kernel<<<12288 / 4, 128, 0, stream>>>(xb, Wqt, Wkt, Wvt, Qb, Kb, Vt);

  // attention: B*H*(T/16) = 8192 waves, 4 waves / block
  flash_kernel<<<8192 / 4, 128, 0, stream>>>(Qb, Kb, Vt, OH);

  // projection: 256 row-tiles(32) * 16 col-groups = 4096 waves, 8 waves / block
  proj_kernel<<<4096 / 8, 256, 0, stream>>>(OH, Wpb, bp, out);
}
